// Block_39539468927393
// MI455X (gfx1250) — matrix operand order repaired and checked by reading, not compiled, not run on hardware
//
#include <hip/hip_runtime.h>
#include <math.h>

// ---------------------------------------------------------------------------
// Transformer block forward for MI455X (gfx1250), bf16 WMMA + fp32 accumulate.
// ~137 GFLOP of matmuls vs ~150MB HBM traffic -> compute bound -> everything
// goes through v_wmma_f32_16x16x32_bf16. Flash attention keeps S/P on-WGP.
// GEMM A-tiles are moved by the Tensor Data Mover (tensor_load_to_lds) where
// the toolchain exposes it, overlapped with cooperative B-tile staging.
// ---------------------------------------------------------------------------

typedef __bf16 bf16;
typedef __bf16 v16bf __attribute__((ext_vector_type(16)));
typedef __bf16 v8bf  __attribute__((ext_vector_type(8)));
typedef float  v8f   __attribute__((ext_vector_type(8)));
typedef unsigned int v4u __attribute__((ext_vector_type(4)));
typedef int  v8i __attribute__((ext_vector_type(8)));
typedef int  v4i __attribute__((ext_vector_type(4)));

#ifndef __has_builtin
#define __has_builtin(x) 0
#endif

#if __has_builtin(__builtin_amdgcn_tensor_load_to_lds) && \
    __has_builtin(__builtin_amdgcn_s_wait_tensorcnt)
#define USE_TDM 1
#else
#define USE_TDM 0
#endif

#define N_EMBD   1024
#define N_HEADS  16
#define HEAD_DIM 64
#define D_FF     4096
#define SEQ      2048
#define BATCH    2
#define MROWS    (BATCH * SEQ)   // 4096 token rows

__device__ __forceinline__ v16bf concat8(v8bf lo, v8bf hi) {
    return __builtin_shufflevector(lo, hi, 0,1,2,3,4,5,6,7,8,9,10,11,12,13,14,15);
}
__device__ __forceinline__ v8f zero8() {
    v8f z;
#pragma unroll
    for (int i = 0; i < 8; ++i) z[i] = 0.0f;
    return z;
}

// ------------------------------ weight cast --------------------------------
__global__ void cast_f32_to_bf16(const float* __restrict__ src,
                                 bf16* __restrict__ dst, int n) {
    int i = blockIdx.x * blockDim.x + threadIdx.x;
    if (i < n) dst[i] = (bf16)src[i];
}

// ------------------------------ layernorm ----------------------------------
// One 256-thread block per 1024-wide row. fp32 math, bf16 out.
__global__ void layernorm_bf16(const float* __restrict__ x,
                               const float* __restrict__ g,
                               const float* __restrict__ b,
                               bf16* __restrict__ out) {
    __shared__ float red[8];
    const int row = blockIdx.x;
    const float* xr = x + (size_t)row * N_EMBD;
    const int tid = threadIdx.x;
    const int wid = tid >> 5, lid = tid & 31;

    float v[4];
    float s = 0.0f;
#pragma unroll
    for (int i = 0; i < 4; ++i) { v[i] = xr[tid + 256 * i]; s += v[i]; }
#pragma unroll
    for (int m = 16; m >= 1; m >>= 1) s += __shfl_xor(s, m, 32);
    if (lid == 0) red[wid] = s;
    __syncthreads();
    float mean = 0.0f;
#pragma unroll
    for (int i = 0; i < 8; ++i) mean += red[i];
    mean *= (1.0f / N_EMBD);
    __syncthreads();

    float q = 0.0f;
#pragma unroll
    for (int i = 0; i < 4; ++i) { float d = v[i] - mean; q += d * d; }
#pragma unroll
    for (int m = 16; m >= 1; m >>= 1) q += __shfl_xor(q, m, 32);
    if (lid == 0) red[wid] = q;
    __syncthreads();
    float var = 0.0f;
#pragma unroll
    for (int i = 0; i < 8; ++i) var += red[i];
    var *= (1.0f / N_EMBD);
    const float rstd = rsqrtf(var + 1e-5f);

#pragma unroll
    for (int i = 0; i < 4; ++i) {
        int c = tid + 256 * i;
        out[(size_t)row * N_EMBD + c] = (bf16)((v[i] - mean) * rstd * g[c] + b[c]);
    }
}

// -------------------------------- GEMM -------------------------------------
// C[M,N] = A[M,K] @ B[K,N], A/B bf16 row-major, fp32 accumulate via WMMA.
// Block: 256 thr (8 waves), tile 64(M) x 128(N), BK=32. Waves: 2(M) x 4(N),
// each wave computes 32x32 = 2x2 WMMA accumulators.
// A tile is DMA'd by the TDM (wave 0 issues tensor_load_to_lds, TDM padding
// reproduces the LDA=48 row stride); B tile is cooperatively transposed.
// MODE 0: out bf16             (Q/K/V projections)
// MODE 1: out f32 = acc + bias[n] + resid[m,n]   (Wo, W2 residual paths)
// MODE 2: out bf16 = gelu(acc + bias[n])         (W1 path, exact GELU)
#define LDA 48    // bf16 elements; 96B rows keep 16B alignment for b128 ops
template <int MODE>
__global__ __launch_bounds__(256, 1)
void gemm_bf16(const bf16* __restrict__ A, const bf16* __restrict__ B,
               const float* __restrict__ bias,
               const float* __restrict__ resid,
               void* __restrict__ outp, int M, int N, int K) {
    __shared__ __align__(16) bf16 ldsA[64 * LDA];
    __shared__ __align__(16) bf16 ldsBt[128 * LDA];   // transposed: [n][k]

    const int tid = threadIdx.x;
    const int wave = tid >> 5, lane = tid & 31;
    const int lane15 = lane & 15, hsel = lane >> 4;   // lane half (0/1)
    const int m0 = blockIdx.y * 64, n0 = blockIdx.x * 128;
    const int wm = (wave & 1) * 32;      // wave M offset within tile
    const int wn = (wave >> 1) * 32;     // wave N offset within tile

    v8f acc[2][2];
#pragma unroll
    for (int i = 0; i < 2; ++i)
#pragma unroll
        for (int j = 0; j < 2; ++j) acc[i][j] = zero8();

#if USE_TDM
    const unsigned ldsA_addr = (unsigned)(unsigned long long)(uintptr_t)&ldsA[0];
#endif

    for (int k0 = 0; k0 < K; k0 += 32) {
        __syncthreads();
#if USE_TDM
        // --- A tile via Tensor Data Mover: 64x32 bf16 2D tile ---
        // D# group0: count=1 | lds_addr | global_addr | type=2
        // D# group1: data_size=2B, pad_enable, pad_interval=16 DW (code 3),
        //            pad_amount=8 DW (code 7) -> LDS row stride 48 bf16 = LDA.
        //            tensor_dim0=K, tensor_dim1=M, tile 32x64, dim0_stride=K.
        if (wave == 0) {
            unsigned long long ga =
                (unsigned long long)(uintptr_t)(A + (size_t)m0 * K + k0);
            v4u g0;
            g0[0] = 1u;                                   // count=1 (valid D#)
            g0[1] = ldsA_addr;                            // lds_addr bytes
            g0[2] = (unsigned)(ga & 0xffffffffu);         // global_addr[31:0]
            g0[3] = (unsigned)((ga >> 32) & 0x01ffffffu)  // global_addr[56:32]
                    | (2u << 30);                         // type=2 ("image")
            unsigned kd = (unsigned)K, md = (unsigned)M;
            v8i g1;
            g1[0] = (int)((1u << 16)      // data_size = 1 -> 2 bytes
                        | (1u << 20)      // pad_enable
                        | (3u << 22)      // pad_interval: 16 DWORDs
                        | (7u << 25));    // pad_amount: 8 DWORDs
            g1[1] = (int)((kd & 0xffffu) << 16);              // dim0[15:0]
            g1[2] = (int)((kd >> 16) | ((md & 0xffffu) << 16));// dim0[31:16],dim1[15:0]
            g1[3] = (int)((md >> 16) | (32u << 16));          // dim1[31:16],tile0=32
            g1[4] = (int)64u;                                 // tile1=64, tile2=0
            g1[5] = (int)kd;                                  // dim0_stride[31:0]
            g1[6] = 0; g1[7] = 0;                             // stride hi, dim1_stride
            v4i gz4 = {0, 0, 0, 0};
#if __clang_major__ >= 23
            v8i gz8 = {0, 0, 0, 0, 0, 0, 0, 0};
            __builtin_amdgcn_tensor_load_to_lds(g0, g1, gz4, gz4, gz8, 0);
#else
            __builtin_amdgcn_tensor_load_to_lds(g0, g1, gz4, gz4, 0);
#endif
        }
#else
        // stage A tile 64x32 (8 contiguous bf16 per thread)
        {
            int r = tid >> 2;
            int c = (tid & 3) * 8;
            const bf16* ap = A + (size_t)(m0 + r) * K + k0 + c;
            v8bf av = *(const v8bf*)ap;
            *(v8bf*)&ldsA[r * LDA + c] = av;
            if (k0 + 32 < K) __builtin_prefetch(ap + 32, 0, 1);  // global_prefetch_b8
        }
#endif
        // stage B tile 32x128 transposed into ldsBt[n][k] (overlaps TDM)
        {
            int g0i = tid * 2;
#pragma unroll
            for (int gi = 0; gi < 2; ++gi) {
                int g = g0i + gi;
                int kr = g >> 4;
                int nc = (g & 15) * 8;
                const bf16* bp = B + (size_t)(k0 + kr) * N + n0 + nc;
                v8bf bv = *(const v8bf*)bp;
                if (k0 + 32 < K) __builtin_prefetch(bp + (size_t)32 * N, 0, 1);
#pragma unroll
                for (int j = 0; j < 8; ++j) ldsBt[(nc + j) * LDA + kr] = bv[j];
            }
        }
#if USE_TDM
        if (wave == 0) __builtin_amdgcn_s_wait_tensorcnt(0);
#endif
        __syncthreads();

        // A fragments: row = lane&15; K elems 0..15 or 16..31
        v16bf afrag[2], bfrag[2];
#pragma unroll
        for (int i = 0; i < 2; ++i) {
            int row = wm + i * 16 + lane15;
            int kb = hsel * 16;
            v8bf lo = *(const v8bf*)&ldsA[row * LDA + kb];
            v8bf hi = *(const v8bf*)&ldsA[row * LDA + kb + 8];
            afrag[i] = concat8(lo, hi);
        }
        // B fragments: col = lane&15; K elems 0..15 or 16..31 (contiguous in Bt)
#pragma unroll
        for (int j = 0; j < 2; ++j) {
            int col = wn + j * 16 + lane15;
            int kb = hsel * 16;
            v8bf lo = *(const v8bf*)&ldsBt[col * LDA + kb];
            v8bf hi = *(const v8bf*)&ldsBt[col * LDA + kb + 8];
            bfrag[j] = concat8(lo, hi);
        }
#pragma unroll
        for (int i = 0; i < 2; ++i)
#pragma unroll
            for (int j = 0; j < 2; ++j)
                acc[i][j] = __builtin_amdgcn_wmma_f32_16x16x32_bf16(
                    false, afrag[i], false, bfrag[j], (short)0, acc[i][j],
                    false, false);
    }

    // epilogue: D layout row = r + 8*hsel, col = lane&15
#pragma unroll
    for (int i = 0; i < 2; ++i)
#pragma unroll
        for (int j = 0; j < 2; ++j) {
            int colg = n0 + wn + j * 16 + lane15;
#pragma unroll
            for (int r = 0; r < 8; ++r) {
                int rowg = m0 + wm + i * 16 + r + hsel * 8;
                size_t idx = (size_t)rowg * N + colg;
                float av = acc[i][j][r];
                if (MODE == 0) {
                    ((bf16*)outp)[idx] = (bf16)av;
                } else if (MODE == 1) {
                    ((float*)outp)[idx] = av + bias[colg] + resid[idx];
                } else {
                    float t = av + bias[colg];
                    float ge = 0.5f * t * (1.0f + erff(t * 0.70710678118654752f));
                    ((bf16*)outp)[idx] = (bf16)ge;
                }
            }
        }
}

// ---------------------------- flash attention ------------------------------
// Grid: B*H*(T/64) blocks, 128 thr (4 waves). Wave w owns 16 query rows.
// Online softmax; S and P tiles never leave the WGP. Scale = 1/sqrt(64).
// K and V tiles are LDS-staged (K row-major IS the B-fragment layout for S,
// V is transposed) so the hot loop only uses ds_load_b128 with immediate
// offsets -- no lane-variant 64-bit global pointers to spill.
#define LDV 72    // 144B rows, 16B aligned
__global__ __launch_bounds__(128, 1)
void attention_fa(const bf16* __restrict__ Q, const bf16* __restrict__ Km,
                  const bf16* __restrict__ V, bf16* __restrict__ O) {
    __shared__ __align__(16) bf16 KT[64 * LDV];          // K tile row-major [n][k]
    __shared__ __align__(16) bf16 VT[64 * LDV];          // V transposed: [n][k]
    __shared__ __align__(16) bf16 Pb[4][16 * LDV];       // per-wave P buffer

    const int tid = threadIdx.x;
    const int wave = tid >> 5, lane = tid & 31;
    const int lane15 = lane & 15, hsel = lane >> 4;
    const int id = blockIdx.x;
    const int qt = id & 31;            // query tile (T/64 = 32)
    const int bh = id >> 5;
    const int h = bh & (N_HEADS - 1);
    const int b = bh >> 4;
    const size_t headoff = (size_t)h * HEAD_DIM;
    const float scale = 0.125f;        // 1/sqrt(HEAD_DIM)

    // Q A-fragments straight from global (contiguous along head dim)
    const int qrow = qt * 64 + wave * 16 + lane15;
    const bf16* qptr = Q + ((size_t)(b * SEQ + qrow)) * N_EMBD + headoff;
    v16bf qa[2];
#pragma unroll
    for (int kd = 0; kd < 2; ++kd) {
        int kb = kd * 32 + hsel * 16;
        v8bf lo = *(const v8bf*)(qptr + kb);
        v8bf hi = *(const v8bf*)(qptr + kb + 8);
        qa[kd] = concat8(lo, hi);
    }

    float mrow[8], lrow[8];
    v8f o[4];
#pragma unroll
    for (int r = 0; r < 8; ++r) { mrow[r] = -__builtin_inff(); lrow[r] = 0.0f; }
#pragma unroll
    for (int nt = 0; nt < 4; ++nt) o[nt] = zero8();

    for (int kt = 0; kt <= qt; ++kt) {
        __syncthreads();               // previous KT/VT/Pb fully consumed
        // stage K tile row-major (32 contiguous bf16 per thread, 4x b128)
        {
            int kr = tid >> 1;
            int nb = (tid & 1) * 32;
            const bf16* kp = Km + ((size_t)(b * SEQ + kt * 64 + kr)) * N_EMBD + headoff + nb;
#pragma unroll
            for (int j = 0; j < 4; ++j)
                *(v8bf*)&KT[kr * LDV + nb + j * 8] = *(const v8bf*)(kp + j * 8);
        }
        // stage V transposed: VT[n][k] = V[kt*64+k][h*64+n]
        {
            int k = tid >> 1;
            int nb = (tid & 1) * 32;
            const bf16* vp = V + ((size_t)(b * SEQ + kt * 64 + k)) * N_EMBD + headoff + nb;
#pragma unroll
            for (int j = 0; j < 32; ++j) VT[(nb + j) * LDV + k] = vp[j];
        }
        __syncthreads();

        // S = Q @ K^T : B-fragment col = key row; contiguous K in KT rows
        v8f s[4];
#pragma unroll
        for (int nt = 0; nt < 4; ++nt) s[nt] = zero8();
#pragma unroll
        for (int nt = 0; nt < 4; ++nt)
#pragma unroll
            for (int kd = 0; kd < 2; ++kd) {
                int n = nt * 16 + lane15;
                int kb = kd * 32 + hsel * 16;
                v8bf lo = *(const v8bf*)&KT[n * LDV + kb];
                v8bf hi = *(const v8bf*)&KT[n * LDV + kb + 8];
                s[nt] = __builtin_amdgcn_wmma_f32_16x16x32_bf16(
                    false, qa[kd], false, concat8(lo, hi), (short)0, s[nt],
                    false, false);
            }

        // scale + causal mask (only diagonal tile needs masking)
        float nm[8];
#pragma unroll
        for (int r = 0; r < 8; ++r) nm[r] = mrow[r];
#pragma unroll
        for (int nt = 0; nt < 4; ++nt)
#pragma unroll
            for (int r = 0; r < 8; ++r) {
                float sv = s[nt][r] * scale;
                if (kt == qt) {
                    int kc = nt * 16 + lane15;
                    int qr = wave * 16 + r + hsel * 8;
                    if (kc > qr) sv = -__builtin_inff();
                }
                s[nt][r] = sv;
                nm[r] = fmaxf(nm[r], sv);
            }
        // row-max across the 16 lanes holding one row (xor stays in half)
#pragma unroll
        for (int r = 0; r < 8; ++r)
#pragma unroll
            for (int mk = 8; mk >= 1; mk >>= 1)
                nm[r] = fmaxf(nm[r], __shfl_xor(nm[r], mk, 32));

        float alpha[8];
#pragma unroll
        for (int r = 0; r < 8; ++r) {
            alpha[r] = expf(mrow[r] - nm[r]);
            mrow[r] = nm[r];
        }
        // P = exp(S - max); row-sum folded into lrow right away; s[] dies here
#pragma unroll
        for (int nt = 0; nt < 4; ++nt)
#pragma unroll
            for (int r = 0; r < 8; ++r) {
                float p = expf(s[nt][r] - nm[r]);
                s[nt][r] = p;
                Pb[wave][(r + 8 * hsel) * LDV + nt * 16 + lane15] = (bf16)p;
            }
#pragma unroll
        for (int r = 0; r < 8; ++r) {
            float rsum = s[0][r] + s[1][r] + s[2][r] + s[3][r];
#pragma unroll
            for (int mk = 8; mk >= 1; mk >>= 1)
                rsum += __shfl_xor(rsum, mk, 32);
            lrow[r] = lrow[r] * alpha[r] + rsum;
        }
#pragma unroll
        for (int nt = 0; nt < 4; ++nt)
#pragma unroll
            for (int r = 0; r < 8; ++r) o[nt][r] *= alpha[r];
        __syncthreads();               // Pb writes visible before reload

        // reload P as A-fragments, O += P @ V (VT gives contiguous B frags)
        v16bf pa[2];
#pragma unroll
        for (int kd = 0; kd < 2; ++kd) {
            int kb = kd * 32 + hsel * 16;
            v8bf lo = *(const v8bf*)&Pb[wave][lane15 * LDV + kb];
            v8bf hi = *(const v8bf*)&Pb[wave][lane15 * LDV + kb + 8];
            pa[kd] = concat8(lo, hi);
        }
#pragma unroll
        for (int nt = 0; nt < 4; ++nt)
#pragma unroll
            for (int kd = 0; kd < 2; ++kd) {
                int n = nt * 16 + lane15;
                int kb = kd * 32 + hsel * 16;
                v8bf lo = *(const v8bf*)&VT[n * LDV + kb];
                v8bf hi = *(const v8bf*)&VT[n * LDV + kb + 8];
                o[nt] = __builtin_amdgcn_wmma_f32_16x16x32_bf16(
                    false, pa[kd], false, concat8(lo, hi), (short)0, o[nt],
                    false, false);
            }
    }

    // normalize and store (bf16, [B,T,C] layout so @Wo is a plain GEMM)
#pragma unroll
    for (int r = 0; r < 8; ++r) {
        float inv = 1.0f / lrow[r];
        int rowg = b * SEQ + qt * 64 + wave * 16 + r + 8 * hsel;
#pragma unroll
        for (int nt = 0; nt < 4; ++nt) {
            size_t idx = (size_t)rowg * N_EMBD + headoff + nt * 16 + lane15;
            O[idx] = (bf16)(o[nt][r] * inv);
        }
    }
}

// ------------------------------- launcher ----------------------------------
extern "C" void kernel_launch(void* const* d_in, const int* in_sizes, int n_in,
                              void* d_out, int out_size, void* d_ws, size_t ws_size,
                              hipStream_t stream) {
    const float* x   = (const float*)d_in[0];
    const float* Wq  = (const float*)d_in[1];
    const float* Wk  = (const float*)d_in[2];
    const float* Wv  = (const float*)d_in[3];
    const float* Wo  = (const float*)d_in[4];
    const float* bo  = (const float*)d_in[5];
    const float* W1  = (const float*)d_in[6];
    const float* b1  = (const float*)d_in[7];
    const float* W2  = (const float*)d_in[8];
    const float* b2  = (const float*)d_in[9];
    const float* g1  = (const float*)d_in[10];
    const float* be1 = (const float*)d_in[11];
    const float* g2  = (const float*)d_in[12];
    const float* be2 = (const float*)d_in[13];

    char* w = (char*)d_ws;
    auto carve = [&](size_t bytes) {
        char* p = w;
        w += (bytes + 255) & ~(size_t)255;
        return p;
    };
    const size_t CC  = (size_t)N_EMBD * N_EMBD;     // 1M
    const size_t CF  = (size_t)N_EMBD * D_FF;       // 4M
    const size_t TOK = (size_t)MROWS * N_EMBD;      // 4M
    bf16* Wq_b = (bf16*)carve(CC * 2);
    bf16* Wk_b = (bf16*)carve(CC * 2);
    bf16* Wv_b = (bf16*)carve(CC * 2);
    bf16* Wo_b = (bf16*)carve(CC * 2);
    bf16* W1_b = (bf16*)carve(CF * 2);
    bf16* W2_b = (bf16*)carve(CF * 2);
    bf16* h_b  = (bf16*)carve(TOK * 2);
    bf16* q_b  = (bf16*)carve(TOK * 2);
    bf16* k_b  = (bf16*)carve(TOK * 2);
    bf16* v_b  = (bf16*)carve(TOK * 2);
    bf16* ao_b = (bf16*)carve(TOK * 2);
    float* x1  = (float*)carve(TOK * 4);
    bf16* h2_b = (bf16*)carve(TOK * 2);
    bf16* m_b  = (bf16*)carve((size_t)MROWS * D_FF * 2);

    // 1) weights -> bf16
    cast_f32_to_bf16<<<(int)((CC + 255) / 256), 256, 0, stream>>>(Wq, Wq_b, (int)CC);
    cast_f32_to_bf16<<<(int)((CC + 255) / 256), 256, 0, stream>>>(Wk, Wk_b, (int)CC);
    cast_f32_to_bf16<<<(int)((CC + 255) / 256), 256, 0, stream>>>(Wv, Wv_b, (int)CC);
    cast_f32_to_bf16<<<(int)((CC + 255) / 256), 256, 0, stream>>>(Wo, Wo_b, (int)CC);
    cast_f32_to_bf16<<<(int)((CF + 255) / 256), 256, 0, stream>>>(W1, W1_b, (int)CF);
    cast_f32_to_bf16<<<(int)((CF + 255) / 256), 256, 0, stream>>>(W2, W2_b, (int)CF);

    // 2) LN1
    layernorm_bf16<<<MROWS, 256, 0, stream>>>(x, g1, be1, h_b);

    // 3) QKV projections
    dim3 gCC(N_EMBD / 128, MROWS / 64);
    gemm_bf16<0><<<gCC, 256, 0, stream>>>(h_b, Wq_b, nullptr, nullptr, q_b, MROWS, N_EMBD, N_EMBD);
    gemm_bf16<0><<<gCC, 256, 0, stream>>>(h_b, Wk_b, nullptr, nullptr, k_b, MROWS, N_EMBD, N_EMBD);
    gemm_bf16<0><<<gCC, 256, 0, stream>>>(h_b, Wv_b, nullptr, nullptr, v_b, MROWS, N_EMBD, N_EMBD);

    // 4) causal flash attention
    attention_fa<<<BATCH * N_HEADS * (SEQ / 64), 128, 0, stream>>>(q_b, k_b, v_b, ao_b);

    // 5) x1 = x + ao@Wo + bo
    gemm_bf16<1><<<gCC, 256, 0, stream>>>(ao_b, Wo_b, bo, x, x1, MROWS, N_EMBD, N_EMBD);

    // 6) LN2
    layernorm_bf16<<<MROWS, 256, 0, stream>>>(x1, g2, be2, h2_b);

    // 7) m = gelu(h2@W1 + b1)
    dim3 gFF(D_FF / 128, MROWS / 64);
    gemm_bf16<2><<<gFF, 256, 0, stream>>>(h2_b, W1_b, b1, nullptr, m_b, MROWS, D_FF, N_EMBD);

    // 8) out = x1 + m@W2 + b2
    gemm_bf16<1><<<gCC, 256, 0, stream>>>(m_b, W2_b, b2, x1, d_out, MROWS, N_EMBD, D_FF);
}
